// CAM_DynamicShape_49495203119666
// MI455X (gfx1250) — matrix operand order repaired and checked by reading, not compiled, not run on hardware
//
#include <hip/hip_runtime.h>
#include <hip/hip_bf16.h>

// CAM (channel attention) for inputs [16,64,64,512] fp32.
//   aTa = A^T A  (512x512x4096 GEMM per batch)  -> ws
//   attn = softmax_rows(aTa)                    -> ws (in place)
//   out  = gamma * (A attn^T) + inputs          (4096x512x512 GEMM per batch)
// GEMMs: v_wmma_f32_16x16x32_bf16, fp32->bf16 on the fly, f32 accumulation.
// LDS tiles are K-contiguous so fragment loads are aligned ds_load_b128.

typedef __attribute__((ext_vector_type(16))) __bf16          v16bf;
typedef __attribute__((ext_vector_type(16))) unsigned short  v16u;
typedef __attribute__((ext_vector_type(8)))  unsigned short  v8u;
typedef __attribute__((ext_vector_type(4)))  unsigned short  v4u;
typedef __attribute__((ext_vector_type(8)))  float           v8f;

#define N_HW 4096
#define C_CH 512
#define KS   72   // LDS row stride in halves: 144 B, multiple of 16 B

__device__ __forceinline__ unsigned short f32_bf16(float f) {
  union { float f; unsigned u; } x; x.f = f;
  unsigned u = x.u + 0x7FFFu + ((x.u >> 16) & 1u);   // round-to-nearest-even
  return (unsigned short)(u >> 16);
}

__device__ __forceinline__ v4u pack4(float a, float b, float c, float d) {
  v4u r;
  r[0] = f32_bf16(a); r[1] = f32_bf16(b);
  r[2] = f32_bf16(c); r[3] = f32_bf16(d);
  return r;
}

// Two aligned 16-byte LDS loads -> one 16-half fragment.
__device__ __forceinline__ v16bf frag16(const unsigned short* p0,
                                        const unsigned short* p1) {
  v8u lo = *(const v8u*)p0;
  v8u hi = *(const v8u*)p1;
  v16u u = __builtin_shufflevector(lo, hi, 0, 1, 2, 3, 4, 5, 6, 7,
                                           8, 9, 10, 11, 12, 13, 14, 15);
  return __builtin_bit_cast(v16bf, u);
}

__device__ __forceinline__ v8f wmma_bf16(v16bf a, v16bf b, v8f c) {
  return __builtin_amdgcn_wmma_f32_16x16x32_bf16(false, a, false, b,
                                                 (short)0, c, false, false);
}

// ---------------------------------------------------------------- gram: A^T A
// Block tile 128(M=c) x 64(N=c); 8 waves in 4x2, each wave 32x32 (4 wmma accs).
__global__ __launch_bounds__(256)
void cam_gram(const float* __restrict__ in, float* __restrict__ aTa) {
  const int b    = blockIdx.z;
  const int cM   = blockIdx.x * 128;
  const int cN   = blockIdx.y * 64;
  const int tid  = threadIdx.x;
  const int lane = tid & 31, wave = tid >> 5;
  const int hi   = lane >> 4, l15 = lane & 15;
  const int wm   = wave & 3, wn = wave >> 2;

  __shared__ __align__(16) unsigned short tM[128][KS]; // [c_local][k(n)]
  __shared__ __align__(16) unsigned short tN[64][KS];  // [c_local][k(n)]

  v8f acc00 = {}, acc01 = {}, acc10 = {}, acc11 = {};
  const float* Ab = in + (size_t)b * N_HW * C_CH;

  const int mr0 = (wm * 2 + 0) * 16 + l15;
  const int mr1 = (wm * 2 + 1) * 16 + l15;
  const int nr0 = (wn * 2 + 0) * 16 + l15;
  const int nr1 = (wn * 2 + 1) * 16 + l15;

  for (int k0 = 0; k0 < N_HW; k0 += 64) {
    // Stage tM: 128c x 64k, transpose on the fly. Coalesced f32 row reads,
    // 4 bf16 packed per aligned ds_store_b64.
#pragma unroll
    for (int p = 0; p < 8; ++p) {
      int cc = tid & 127;
      int kk = p * 8 + (tid >> 7) * 4;
      const float* base = Ab + (size_t)(k0 + kk) * C_CH + cM + cc;
      *(v4u*)&tM[cc][kk] =
          pack4(base[0], base[C_CH], base[2 * C_CH], base[3 * C_CH]);
    }
    // Stage tN: 64c x 64k.
#pragma unroll
    for (int p = 0; p < 4; ++p) {
      int cc = tid & 63;
      int kk = p * 16 + (tid >> 6) * 4;
      const float* base = Ab + (size_t)(k0 + kk) * C_CH + cN + cc;
      *(v4u*)&tN[cc][kk] =
          pack4(base[0], base[C_CH], base[2 * C_CH], base[3 * C_CH]);
    }
    __syncthreads();

#pragma unroll
    for (int ks = 0; ks < 64; ks += 32) {
      v16bf a0 = frag16(&tM[mr0][ks + hi * 8], &tM[mr0][ks + 16 + hi * 8]);
      v16bf a1 = frag16(&tM[mr1][ks + hi * 8], &tM[mr1][ks + 16 + hi * 8]);
      v16bf b0 = frag16(&tN[nr0][ks + hi * 8], &tN[nr0][ks + 16 + hi * 8]);
      v16bf b1 = frag16(&tN[nr1][ks + hi * 8], &tN[nr1][ks + 16 + hi * 8]);
      acc00 = wmma_bf16(a0, b0, acc00);
      acc01 = wmma_bf16(a0, b1, acc01);
      acc10 = wmma_bf16(a1, b0, acc10);
      acc11 = wmma_bf16(a1, b1, acc11);
    }
    __syncthreads();
  }

  float* O = aTa + (size_t)b * C_CH * C_CH;
#pragma unroll
  for (int r = 0; r < 8; ++r) {  // C/D layout: M = r + 8*hi, N = l15
    int m0 = cM + wm * 32 + r + hi * 8;
    int m1 = m0 + 16;
    int n0 = cN + wn * 32 + l15;
    int n1 = n0 + 16;
    O[(size_t)m0 * C_CH + n0] = acc00[r];
    O[(size_t)m0 * C_CH + n1] = acc01[r];
    O[(size_t)m1 * C_CH + n0] = acc10[r];
    O[(size_t)m1 * C_CH + n1] = acc11[r];
  }
}

// ------------------------------------------------------------ softmax (rows)
__global__ __launch_bounds__(256)
void cam_softmax(float* __restrict__ aTa) {
  float* p = aTa + (size_t)blockIdx.x * C_CH;
  const int t = threadIdx.x;
  __shared__ float red[256];

  float v0 = p[t], v1 = p[t + 256];
  red[t] = fmaxf(v0, v1);
  __syncthreads();
#pragma unroll
  for (int s = 128; s > 0; s >>= 1) {
    if (t < s) red[t] = fmaxf(red[t], red[t + s]);
    __syncthreads();
  }
  float rmax = red[0];
  __syncthreads();

  float e0 = __expf(v0 - rmax), e1 = __expf(v1 - rmax);
  red[t] = e0 + e1;
  __syncthreads();
#pragma unroll
  for (int s = 128; s > 0; s >>= 1) {
    if (t < s) red[t] += red[t + s];
    __syncthreads();
  }
  float inv = 1.0f / red[0];
  p[t] = e0 * inv;
  p[t + 256] = e1 * inv;
}

// ------------------------------------------- apply: out = g*(A attn^T) + in
// Block tile 128(M=hw) x 64(N=c); K = channel is contiguous in memory for
// BOTH operands -> no transpose: global_load_b128 -> pack -> ds_store_b64.
__global__ __launch_bounds__(256)
void cam_apply(const float* __restrict__ in, const float* __restrict__ attn,
               const float* __restrict__ gamma, float* __restrict__ out) {
  const int b    = blockIdx.z;
  const int nM   = blockIdx.x * 128;
  const int cN   = blockIdx.y * 64;
  const int tid  = threadIdx.x;
  const int lane = tid & 31, wave = tid >> 5;
  const int hi   = lane >> 4, l15 = lane & 15;
  const int wm   = wave & 3, wn = wave >> 2;

  __shared__ __align__(16) unsigned short tA[128][KS]; // [hw_local][k(d)]
  __shared__ __align__(16) unsigned short tB[64][KS];  // [c_local][k(d)]

  v8f acc00 = {}, acc01 = {}, acc10 = {}, acc11 = {};
  const float* Ab = in   + (size_t)b * N_HW * C_CH;
  const float* Wb = attn + (size_t)b * C_CH * C_CH;

  const int mr0 = (wm * 2 + 0) * 16 + l15;
  const int mr1 = (wm * 2 + 1) * 16 + l15;
  const int nr0 = (wn * 2 + 0) * 16 + l15;
  const int nr1 = (wn * 2 + 1) * 16 + l15;

  for (int k0 = 0; k0 < C_CH; k0 += 64) {
#pragma unroll
    for (int p = 0; p < 8; ++p) {
      int mm = tid >> 1;
      int kk = p * 8 + (tid & 1) * 4;
      const float* src = Ab + (size_t)(nM + mm) * C_CH + k0 + kk;
      float4 f = *(const float4*)src;
      *(v4u*)&tA[mm][kk] = pack4(f.x, f.y, f.z, f.w);
      if (k0 + 64 < C_CH && p == 0)
        __builtin_prefetch(src + 64, 0, 0);   // global_prefetch_b8 next chunk
    }
#pragma unroll
    for (int p = 0; p < 4; ++p) {
      int cc = tid >> 2;
      int kk = p * 16 + (tid & 3) * 4;
      const float* src = Wb + (size_t)(cN + cc) * C_CH + k0 + kk;
      float4 f = *(const float4*)src;
      *(v4u*)&tB[cc][kk] = pack4(f.x, f.y, f.z, f.w);
    }
    __syncthreads();

#pragma unroll
    for (int ks = 0; ks < 64; ks += 32) {
      v16bf a0 = frag16(&tA[mr0][ks + hi * 8], &tA[mr0][ks + 16 + hi * 8]);
      v16bf a1 = frag16(&tA[mr1][ks + hi * 8], &tA[mr1][ks + 16 + hi * 8]);
      v16bf b0 = frag16(&tB[nr0][ks + hi * 8], &tB[nr0][ks + 16 + hi * 8]);
      v16bf b1 = frag16(&tB[nr1][ks + hi * 8], &tB[nr1][ks + 16 + hi * 8]);
      acc00 = wmma_bf16(a0, b0, acc00);
      acc01 = wmma_bf16(a0, b1, acc01);
      acc10 = wmma_bf16(a1, b0, acc10);
      acc11 = wmma_bf16(a1, b1, acc11);
    }
    __syncthreads();
  }

  const float g = gamma[0];
  float* Ob = out + (size_t)b * N_HW * C_CH;
#pragma unroll
  for (int r = 0; r < 8; ++r) {
    int m0 = nM + wm * 32 + r + hi * 8;
    int m1 = m0 + 16;
    int n0 = cN + wn * 32 + l15;
    int n1 = n0 + 16;
    size_t i00 = (size_t)m0 * C_CH + n0, i01 = (size_t)m0 * C_CH + n1;
    size_t i10 = (size_t)m1 * C_CH + n0, i11 = (size_t)m1 * C_CH + n1;
    Ob[i00] = g * acc00[r] + Ab[i00];
    Ob[i01] = g * acc01[r] + Ab[i01];
    Ob[i10] = g * acc10[r] + Ab[i10];
    Ob[i11] = g * acc11[r] + Ab[i11];
  }
}

extern "C" void kernel_launch(void* const* d_in, const int* in_sizes, int n_in,
                              void* d_out, int out_size, void* d_ws, size_t ws_size,
                              hipStream_t stream) {
  const float* in    = (const float*)d_in[0];   // [16,64,64,512] fp32
  const float* gamma = (const float*)d_in[1];   // [1] fp32
  float* out = (float*)d_out;                   // [16,64,64,512] fp32
  float* aTa = (float*)d_ws;                    // 16*512*512*4 = 16 MB scratch

  dim3 blk(256, 1, 1);
  dim3 g1(C_CH / 128, C_CH / 64, 16);           // 4 x 8 x 16
  cam_gram<<<g1, blk, 0, stream>>>(in, aTa);

  cam_softmax<<<dim3(16 * C_CH, 1, 1), blk, 0, stream>>>(aTa);

  dim3 g3(N_HW / 128, C_CH / 64, 16);           // 32 x 8 x 16
  cam_apply<<<g3, blk, 0, stream>>>(in, aTa, gamma, out);
}
